// QuantumEnhancedFlow_34583076667602
// MI455X (gfx1250) — compile-verified, matrix operand order checked by reading, not run on hardware
//
#include <hip/hip_runtime.h>
#include <hip/hip_bf16.h>

// ---------------------------------------------------------------------------
// QuantumEnhancedFlow on MI455X (gfx1250, wave32)
//   Stage 1: 10-qubit statevector sim, one state per wave, state in LDS
//   Stage 2: MLP 64->512->512->512->128 via v_wmma_f32_16x16x32_f16
//            (64x64 register-blocked wave tile; block-shared W tile staged
//             into LDS in fragment order via GLOBAL_LOAD_ASYNC_TO_LDS_B128)
//   Stage 3: per-row log-prob reduction (wave per row)
// ---------------------------------------------------------------------------

typedef __attribute__((ext_vector_type(16))) _Float16 v16h;
typedef __attribute__((ext_vector_type(8)))  _Float16 v8h;
typedef __attribute__((ext_vector_type(8)))  float    v8f;
typedef __attribute__((ext_vector_type(4)))  int      i32x4;

#define BATCH   32768
#define NQ      10
#define QDIM    1024
#define FEAT    64
#define HID     512
#define OUTF    128
#define LOG2PI  1.8378770664093453f

#ifndef __has_builtin
#define __has_builtin(x) 0
#endif
#if defined(__HIP_DEVICE_COMPILE__) && \
    __has_builtin(__builtin_amdgcn_global_load_async_to_lds_b128) && \
    __has_builtin(__builtin_amdgcn_s_wait_asynccnt)
#define USE_ASYNC_LDS 1
#else
#define USE_ASYNC_LDS 0
#endif

// ---------------------------------------------------------------------------
// small complex helper
// ---------------------------------------------------------------------------
struct cplx { float re, im; };
__device__ __forceinline__ cplx cmul(cplx a, cplx b) {
    return { a.re * b.re - a.im * b.im, a.re * b.im + a.im * b.re };
}

// ---------------------------------------------------------------------------
// f32 -> f16 weight conversion
// ---------------------------------------------------------------------------
__global__ void f32_to_f16_kernel(const float* __restrict__ a,
                                  _Float16* __restrict__ b, int n) {
    int i = blockIdx.x * blockDim.x + threadIdx.x;
    if (i < n) b[i] = (_Float16)a[i];
}

// ---------------------------------------------------------------------------
// Quantum expectations. Block = 256 threads = 8 waves; each wave owns one
// batch element's 1024-amp state in LDS (8 x 8KB = 64KB << 320KB/WGP).
// Writes "enhanced" = concat(q[10], x[10:64]) as both f16 (GEMM A) and f32.
// ---------------------------------------------------------------------------
__global__ __launch_bounds__(256)
void quantum_kernel(const float* __restrict__ x, const float* __restrict__ theta,
                    _Float16* __restrict__ enh16, float* __restrict__ enh32) {
    __shared__ float2 st[8 * QDIM];   // 64 KB
    const int lane = threadIdx.x & 31;
    const int wv   = threadIdx.x >> 5;
    const int elem = blockIdx.x * 8 + wv;
    float2* s = st + wv * QDIM;
    const float* xr = x + (size_t)elem * FEAT;

    // ---- encoding layer collapsed: Ry(2*atan(x_q)) on |0..0> -> product state
    float cq[NQ], sq[NQ];
#pragma unroll
    for (int q = 0; q < NQ; ++q) {
        float xv = xr[q];
        float c = rsqrtf(1.0f + xv * xv);   // cos(atan(x)) = 1/sqrt(1+x^2)
        cq[q] = c; sq[q] = xv * c;          // sin(atan(x)) = x/sqrt(1+x^2)
    }
    for (int j = lane; j < QDIM; j += 32) {
        float amp = 1.0f;
#pragma unroll
        for (int q = 0; q < NQ; ++q) amp *= ((j >> q) & 1) ? sq[q] : cq[q];
        s[j] = make_float2(amp, 0.0f);
    }
    __syncthreads();

    // ---- parameterized layers: fused U = Rz * Ry * Rx per qubit, then CX ring
    for (int d = 0; d < 2; ++d) {
        for (int q = 0; q < NQ; ++q) {
            const int base = (d * NQ + q) * 3;
            float sx, cx, sy, cy, sz, cz;
            sincosf(0.5f * theta[base + 0], &sx, &cx);
            sincosf(0.5f * theta[base + 1], &sy, &cy);
            sincosf(0.5f * theta[base + 2], &sz, &cz);
            // M = Ry*Rx ; U = Rz*M
            cplx m00 = {  cy * cx,  sy * sx };
            cplx m01 = { -sy * cx, -cy * sx };
            cplx m10 = {  sy * cx, -cy * sx };
            cplx m11 = {  cy * cx, -sy * sx };
            cplx ezm = { cz, -sz }, ezp = { cz, sz };
            cplx u00 = cmul(ezm, m00), u01 = cmul(ezm, m01);
            cplx u10 = cmul(ezp, m10), u11 = cmul(ezp, m11);
            const int lowmask = (1 << q) - 1;
            for (int t = lane; t < QDIM / 2; t += 32) {
                int j0 = ((t & ~lowmask) << 1) | (t & lowmask); // insert 0 at bit q
                int j1 = j0 | (1 << q);
                float2 a0 = s[j0], a1 = s[j1];
                float2 r0, r1;
                r0.x = u00.re * a0.x - u00.im * a0.y + u01.re * a1.x - u01.im * a1.y;
                r0.y = u00.re * a0.y + u00.im * a0.x + u01.re * a1.y + u01.im * a1.x;
                r1.x = u10.re * a0.x - u10.im * a0.y + u11.re * a1.x - u11.im * a1.y;
                r1.y = u10.re * a0.y + u10.im * a0.x + u11.re * a1.y + u11.im * a1.x;
                s[j0] = r0; s[j1] = r1;
            }
            __syncthreads();
        }
        // CX ring permutations: swap amps where bit_c==1, bit_t==0
        for (int p = 0; p < NQ; ++p) {
            const int c = p, t = (p < NQ - 1) ? p + 1 : 0;
            const int lo = c < t ? c : t, hi = c < t ? t : c;
            for (int i = lane; i < QDIM / 4; i += 32) {
                int j = i;
                j = ((j >> lo) << (lo + 1)) | (j & ((1 << lo) - 1));
                j = ((j >> hi) << (hi + 1)) | (j & ((1 << hi) - 1));
                j |= (1 << c);                 // bit_c = 1, bit_t = 0
                int j2 = j | (1 << t);
                float2 a = s[j], b = s[j2];
                s[j] = b; s[j2] = a;
            }
            __syncthreads();
        }
    }

    // ---- expectations q_k = sum_j |amp_j|^2 * (1 - 2*bit_k(j))
    float acc[NQ];
#pragma unroll
    for (int k = 0; k < NQ; ++k) acc[k] = 0.0f;
    for (int j = lane; j < QDIM; j += 32) {
        float2 a = s[j];
        float pr = a.x * a.x + a.y * a.y;
#pragma unroll
        for (int k = 0; k < NQ; ++k) acc[k] += ((j >> k) & 1) ? -pr : pr;
    }
#pragma unroll
    for (int k = 0; k < NQ; ++k)
#pragma unroll
        for (int m = 16; m >= 1; m >>= 1)
            acc[k] += __shfl_xor(acc[k], m, 32);

    float*    e32 = enh32 + (size_t)elem * FEAT;
    _Float16* e16 = enh16 + (size_t)elem * FEAT;
    if (lane == 0) {
#pragma unroll
        for (int k = 0; k < NQ; ++k) {
            e32[k] = acc[k];
            e16[k] = (_Float16)acc[k];
        }
    }
    for (int f = NQ + lane; f < FEAT; f += 32) {
        float v = xr[f];
        e32[f] = v; e16[f] = (_Float16)v;
    }
}

// ---------------------------------------------------------------------------
// WMMA fragment loader from global (16-bit A layout, ISA 7.12.2):
//   lane<16 : row m, K in {k0..k0+7} U {k0+16..k0+23}
//   lane>=16: row m, K in {k0+8..k0+15} U {k0+24..k0+31}
// -> two contiguous 16-byte loads per lane.
// ---------------------------------------------------------------------------
__device__ __forceinline__ v16h load_frag16(const _Float16* __restrict__ row_base,
                                            int k0, int hi8) {
    const _Float16* p = row_base + k0 + hi8;
    v8h lo = *reinterpret_cast<const v8h*>(p);
    v8h hh = *reinterpret_cast<const v8h*>(p + 16);
    v16h r;
#pragma unroll
    for (int i = 0; i < 8; ++i) { r[i] = lo[i]; r[i + 8] = hh[i]; }
    return r;
}

// ---------------------------------------------------------------------------
// GEMM: C[M,N] = act(A[M,K] @ W[N,K]^T + bias[N])
// Block = 128 threads = 4 waves; block tile 256(M) x 64(N); wave tile 64x64
// as 4x4 WMMA fragments. All 4 waves share W rows [n0, n0+64), so the W tile
// (64 x K <= 64KB) is staged once into LDS in *fragment order*: chunk
// (kstep s, ntile ni) holds each lane's 32B fragment at lane*32 — per-lane
// async gather/scatter exactly matches GLOBAL_LOAD_ASYNC_TO_LDS_B128
// semantics (ASYNCcnt-tracked), and inner-loop B reads become contiguous
// ds_load_b128. Inner loop: 8 global b128 (A) + 8 ds b128 (B) per 16 WMMA.
// Grid divides all dims exactly: no guards, EXEC all-1s for WMMA.
// ---------------------------------------------------------------------------
__global__ __launch_bounds__(128)
void gemm_wmma(const _Float16* __restrict__ A, const _Float16* __restrict__ W,
               const float* __restrict__ bias, _Float16* __restrict__ out16,
               float* __restrict__ out32, int M, int N, int K, int act) {
    __shared__ _Float16 wtile[HID * 64];               // 64 KB max (K<=512)
    const int lane  = threadIdx.x & 31;
    const int wv    = threadIdx.x >> 5;
    const int n0    = blockIdx.x * 64;
    const int m0    = (blockIdx.y * 4 + wv) * 64;
    const int lhalf = lane & 15;
    const int hi8   = (lane >> 4) * 8;

    // ---- stage W[n0:n0+64, 0:K] into LDS, fragment order --------------------
    // (s, ni, piece): lane copies 16B from W[n0+16ni+lhalf][s*32+hi8+16p]
    //                 to   wtile[(s*4+ni)*512 + lane*16 + p*8]   (halfs)
    {
        const int tot = (K >> 5) * 8;                  // (K/32) * 4 ntiles * 2 pieces
        for (int idx = wv; idx < tot; idx += 4) {
            const int s  = idx >> 3;
            const int ni = (idx >> 1) & 3;
            const int p  = idx & 1;
            const _Float16* g = W + (size_t)(n0 + ni * 16 + lhalf) * K
                                  + (s << 5) + hi8 + p * 16;
            _Float16* l = wtile + ((s * 4 + ni) * 512 + lane * 16 + p * 8);
#if USE_ASYNC_LDS
            __builtin_amdgcn_global_load_async_to_lds_b128(
                (__attribute__((address_space(1))) i32x4*)g,
                (__attribute__((address_space(3))) i32x4*)l,
                /*offset=*/0, /*cpol=*/0);
#else
            *reinterpret_cast<v8h*>(l) = *reinterpret_cast<const v8h*>(g);
#endif
        }
#if USE_ASYNC_LDS
        __builtin_amdgcn_s_wait_asynccnt(0);
#endif
        __syncthreads();
    }

    const _Float16* arow = A + (size_t)(m0 + lhalf) * K;
    const size_t rstep = (size_t)16 * K;               // 16 rows

    v8f c[4][4];
#pragma unroll
    for (int mi = 0; mi < 4; ++mi)
#pragma unroll
        for (int ni = 0; ni < 4; ++ni) c[mi][ni] = (v8f){};

    for (int k0 = 0; k0 < K; k0 += 32) {
        const int s = k0 >> 5;
        v16h a[4], b[4];
#pragma unroll
        for (int i = 0; i < 4; ++i) {
            __builtin_prefetch(arow + i * rstep + k0 + 64, 0, 1);
            a[i] = load_frag16(arow + i * rstep, k0, hi8);
            // B fragment from LDS: contiguous 32B per lane
            const _Float16* lb = wtile + ((s * 4 + i) * 512 + lane * 16);
            v8h lo = *reinterpret_cast<const v8h*>(lb);
            v8h hh = *reinterpret_cast<const v8h*>(lb + 8);
#pragma unroll
            for (int e = 0; e < 8; ++e) { b[i][e] = lo[e]; b[i][e + 8] = hh[e]; }
        }
#pragma unroll
        for (int mi = 0; mi < 4; ++mi)
#pragma unroll
            for (int ni = 0; ni < 4; ++ni)
                c[mi][ni] = __builtin_amdgcn_wmma_f32_16x16x32_f16(
                    /*neg_a=*/false, a[mi], /*neg_b=*/false, b[ni],
                    /*c_mod=*/(short)0, c[mi][ni],
                    /*reuse_a=*/false, /*reuse_b=*/false);
    }

    // C/D layout: lane l, VGPR r -> (m0 + 16*mi + r + 8*(l>=16), n0 + 16*ni + l%16)
#pragma unroll
    for (int ni = 0; ni < 4; ++ni) {
        const int n  = n0 + ni * 16 + lhalf;
        const float bn = bias[n];
#pragma unroll
        for (int mi = 0; mi < 4; ++mi) {
#pragma unroll
            for (int r = 0; r < 8; ++r) {
                const int m = m0 + mi * 16 + r + hi8;
                float v = c[mi][ni][r] + bn;
                if (act) v = tanhf(v);
                if (out16) out16[(size_t)m * N + n] = (_Float16)v;
                if (out32) out32[(size_t)m * N + n] = v;
            }
        }
    }
}

// ---------------------------------------------------------------------------
// log_prob[i] = sum_f( -0.5*z^2 - 0.5*log(2pi) - log_scale )
//   z = (enh - shift) * exp(-log_scale); out[:, :64]=shift, out[:, 64:]=ls
// One wave per row, shuffle reduce (wave32).
// ---------------------------------------------------------------------------
__global__ __launch_bounds__(256)
void logprob_kernel(const float* __restrict__ enh, const float* __restrict__ o,
                    float* __restrict__ lp, int M) {
    const int lane = threadIdx.x & 31;
    const int row  = blockIdx.x * 8 + (threadIdx.x >> 5);
    if (row >= M) return;
    const float* e  = enh + (size_t)row * FEAT;
    const float* so = o   + (size_t)row * OUTF;
    float acc = 0.0f;
    for (int f = lane; f < FEAT; f += 32) {
        float shift = so[f];
        float ls    = so[FEAT + f];
        float z     = (e[f] - shift) * __expf(-ls);
        acc += -0.5f * z * z - 0.5f * LOG2PI - ls;
    }
#pragma unroll
    for (int m = 16; m >= 1; m >>= 1) acc += __shfl_xor(acc, m, 32);
    if (lane == 0) lp[row] = acc;
}

// ---------------------------------------------------------------------------
// Host launcher
// ---------------------------------------------------------------------------
extern "C" void kernel_launch(void* const* d_in, const int* in_sizes, int n_in,
                              void* d_out, int out_size, void* d_ws, size_t ws_size,
                              hipStream_t stream) {
    (void)in_sizes; (void)n_in; (void)out_size; (void)ws_size;
    const float* x     = (const float*)d_in[0];
    const float* theta = (const float*)d_in[1];
    const float* W_in  = (const float*)d_in[2];
    const float* b_in  = (const float*)d_in[3];
    const float* W_h   = (const float*)d_in[4];
    const float* b_h   = (const float*)d_in[5];
    const float* W_out = (const float*)d_in[6];
    const float* b_out = (const float*)d_in[7];
    float* lp = (float*)d_out;

    // workspace carve-up (256B aligned)
    char* ws = (char*)d_ws;
    size_t off = 0;
    auto carve = [&](size_t bytes) -> char* {
        char* p = ws + off;
        off += (bytes + 255) & ~size_t(255);
        return p;
    };
    _Float16* enh16  = (_Float16*)carve((size_t)BATCH * FEAT * 2);
    float*    enh32  = (float*)   carve((size_t)BATCH * FEAT * 4);
    _Float16* h0     = (_Float16*)carve((size_t)BATCH * HID  * 2);
    _Float16* h1     = (_Float16*)carve((size_t)BATCH * HID  * 2);
    float*    o32    = (float*)   carve((size_t)BATCH * OUTF * 4);
    _Float16* w16in  = (_Float16*)carve((size_t)HID  * FEAT * 2);
    _Float16* w16h   = (_Float16*)carve((size_t)2 * HID * HID * 2);
    _Float16* w16out = (_Float16*)carve((size_t)OUTF * HID  * 2);

    // weights -> f16
    {
        int n;
        n = HID * FEAT;
        f32_to_f16_kernel<<<(n + 255) / 256, 256, 0, stream>>>(W_in, w16in, n);
        n = 2 * HID * HID;
        f32_to_f16_kernel<<<(n + 255) / 256, 256, 0, stream>>>(W_h, w16h, n);
        n = OUTF * HID;
        f32_to_f16_kernel<<<(n + 255) / 256, 256, 0, stream>>>(W_out, w16out, n);
    }

    // quantum feature map
    quantum_kernel<<<BATCH / 8, 256, 0, stream>>>(x, theta, enh16, enh32);

    // MLP (ping-pong h0/h1); wave tile 64x64, block tile 256x64
    gemm_wmma<<<dim3(HID / 64, BATCH / 256), 128, 0, stream>>>(
        enh16, w16in, b_in, h0, nullptr, BATCH, HID, FEAT, 1);
    gemm_wmma<<<dim3(HID / 64, BATCH / 256), 128, 0, stream>>>(
        h0, w16h, b_h, h1, nullptr, BATCH, HID, HID, 1);
    gemm_wmma<<<dim3(HID / 64, BATCH / 256), 128, 0, stream>>>(
        h1, w16h + (size_t)HID * HID, b_h + HID, h0, nullptr, BATCH, HID, HID, 1);
    gemm_wmma<<<dim3(OUTF / 64, BATCH / 256), 128, 0, stream>>>(
        h0, w16out, b_out, nullptr, o32, BATCH, OUTF, HID, 0);

    // final reduction
    logprob_kernel<<<BATCH / 8, 256, 0, stream>>>(enh32, o32, lp, BATCH);
}
